// MultiheadDiffAttn_26362509262985
// MI455X (gfx1250) — compile-verified, matrix operand order checked by reading, not run on hardware
//
#include <hip/hip_runtime.h>

// ---------------------------------------------------------------------------
// Differential Multi-head Attention for MI455X (gfx1250, wave32, WMMA + TDM)
//   E=2048, H=16, HD=64, b=2, t=2048
// GEMMs + attention matmuls: bf16 WMMA (v_wmma_f32_16x16x32_bf16), fp32 accum.
// All global->LDS tile staging via Tensor Data Mover (tensor_load_to_lds,
// TENSORcnt), double-buffered in the GEMM.
// ---------------------------------------------------------------------------

typedef __bf16 bf16;
typedef __attribute__((ext_vector_type(16))) __bf16 v16bf;
typedef __attribute__((ext_vector_type(8)))  float  v8f;
typedef unsigned int u32x4 __attribute__((ext_vector_type(4)));
typedef int          i32x4 __attribute__((ext_vector_type(4)));
typedef int          i32x8 __attribute__((ext_vector_type(8)));

#define B_  2
#define T_  2048
#define E_  2048
#define H_  16
#define HD_ 64
#define MT_ 4096              // B_*T_
#define LAMBDA_INIT_F      0.78360576653f
#define ONE_MINUS_LAMBDA_F 0.21639423347f

// --- WMMA helper ------------------------------------------------------------
static __device__ __forceinline__ v8f wmma_bf16(v16bf a, v16bf b, v8f c) {
  return __builtin_amdgcn_wmma_f32_16x16x32_bf16(false, a, false, b,
                                                 (short)0, c, false, false);
}

// Load a 16x32 bf16 A/B fragment (CDNA5 layout, ISA 7.12.2):
//   p = base + row(lane&15)*stride ; elements K = 8*half+0..7, 16+8*half+0..7
static __device__ __forceinline__ v16bf load_frag(const bf16* p, int half) {
  union { uint4 u[2]; v16bf v; } f;
  f.u[0] = *(const uint4*)(p + 8 * half);
  f.u[1] = *(const uint4*)(p + 16 + 8 * half);
  return f.v;
}

static __device__ __forceinline__ v8f zero8() {
  v8f z = {0.f,0.f,0.f,0.f,0.f,0.f,0.f,0.f};
  return z;
}

// Generic LDS pointer -> LDS byte offset (flat aperture keeps it in addr[31:0])
static __device__ __forceinline__ unsigned lds_off(const void* p) {
  return (unsigned)(uintptr_t)p;
}

// --- Tensor Data Mover: 2-D bf16 tile load, global -> LDS -------------------
// Tile: tile_x contiguous elements per row, tile_y rows, row stride
// stride_elems; destination packed row-major at lds_byte.
// D# layout per CDNA5 ISA.
static __device__ __forceinline__ void tdm_load_2d(const void* gptr,
                                                   unsigned lds_byte,
                                                   int tile_x, int tile_y,
                                                   int stride_elems) {
  unsigned long long ga = (unsigned long long)(uintptr_t)gptr;
  u32x4 g0;
  g0[0] = 1u;                                           // count=1 (valid D#)
  g0[1] = lds_byte;                                     // lds_addr
  g0[2] = (unsigned)(ga & 0xFFFFFFFFull);               // global_addr[31:0]
  g0[3] = (unsigned)((ga >> 32) & 0x01FFFFFFull)        // global_addr[56:32]
        | (2u << 30);                                   // type = 2 ("image")
  i32x8 g1;
  g1[0] = 1 << 16;                                      // data_size=1 (2 bytes)
  g1[1] = (stride_elems & 0xFFFF) << 16;                // tensor_dim0[15:0]
  g1[2] = ((stride_elems >> 16) & 0xFFFF)               // tensor_dim0[31:16]
        | ((tile_y & 0xFFFF) << 16);                    // tensor_dim1[15:0]
  g1[3] = ((tile_y >> 16) & 0xFFFF)                     // tensor_dim1[31:16]
        | ((tile_x & 0xFFFF) << 16);                    // tile_dim0
  g1[4] = tile_y & 0xFFFF;                              // tile_dim1 (dim2=0)
  g1[5] = stride_elems;                                 // tensor_dim0_stride lo
  g1[6] = 0;                                            // stride hi | dim1_stride
  g1[7] = 0;
  i32x4 z4 = {0, 0, 0, 0};
#if defined(__clang_major__) && (__clang_major__ >= 23)
  i32x8 z8 = {0, 0, 0, 0, 0, 0, 0, 0};
  __builtin_amdgcn_tensor_load_to_lds(g0, g1, z4, z4, z8, 0);
#else
  __builtin_amdgcn_tensor_load_to_lds(g0, g1, z4, z4, 0);
#endif
}

static __device__ __forceinline__ void wait_tensorcnt0() {
  __builtin_amdgcn_s_wait_tensorcnt(0);
}

// --- fp32 -> bf16 elementwise ----------------------------------------------
__global__ void cvt_f32_bf16_kernel(const float* __restrict__ in,
                                    bf16* __restrict__ out, int n) {
  int i = (blockIdx.x * 256 + threadIdx.x) * 4;
  if (i < n) {
    float4 v = *(const float4*)(in + i);
    union { bf16 h[4]; uint2 u; } o;
    o.h[0] = (bf16)v.x; o.h[1] = (bf16)v.y;
    o.h[2] = (bf16)v.z; o.h[3] = (bf16)v.w;
    *(uint2*)(out + i) = o.u;
  }
}

// --- W [K][N] fp32  ->  Wt [N][K] bf16 (transpose+convert) ------------------
__global__ void transpose_cvt_kernel(const float* __restrict__ W,
                                     bf16* __restrict__ Wt, int dim) {
  __shared__ float tile[32][33];
  int bx = blockIdx.x * 32, by = blockIdx.y * 32;
  int tx = threadIdx.x, ty = threadIdx.y;
#pragma unroll
  for (int j = 0; j < 32; j += 8)
    tile[ty + j][tx] = W[(size_t)(by + ty + j) * dim + bx + tx];
  __syncthreads();
#pragma unroll
  for (int j = 0; j < 32; j += 8)
    Wt[(size_t)(bx + ty + j) * dim + by + tx] = (bf16)tile[tx][ty + j];
}

// --- lambda = exp(lq1.lk1) - exp(lq2.lk2) + lambda_init ---------------------
__global__ void lambda_kernel(const float* __restrict__ lq1,
                              const float* __restrict__ lk1,
                              const float* __restrict__ lq2,
                              const float* __restrict__ lk2,
                              float* __restrict__ lam) {
  int i = threadIdx.x;                // 32 threads, HD=64
  float s1 = lq1[i] * lk1[i] + lq1[i + 32] * lk1[i + 32];
  float s2 = lq2[i] * lk2[i] + lq2[i + 32] * lk2[i + 32];
#pragma unroll
  for (int xm = 1; xm < 32; xm <<= 1) {
    s1 += __shfl_xor(s1, xm, 32);
    s2 += __shfl_xor(s2, xm, 32);
  }
  if (i == 0) *lam = __expf(s1) - __expf(s2) + LAMBDA_INIT_F;
}

// --- Tiled bf16 WMMA GEMM: C[M,N] = scale * (A[M,K] @ Bt[N,K]^T) ------------
// Double-buffered LDS tiles filled by the Tensor Data Mover (wave 0 issues,
// everyone waits TENSORcnt). One barrier per 32-K chunk.
// OUT_MODE 0: bf16 row-major   1: bf16 transposed per batch (for V^T)
// OUT_MODE 2: fp32 row-major (final output)
template <int OUT_MODE>
__global__ void __launch_bounds__(256)
gemm_bf16_kernel(const bf16* __restrict__ A, const bf16* __restrict__ Bt,
                 void* __restrict__ Cout, int N, int K, float scale) {
  __shared__ __attribute__((aligned(16))) bf16 sA[2][128 * 32];
  __shared__ __attribute__((aligned(16))) bf16 sB[2][128 * 32];

  const int tid = threadIdx.x;
  const int wave = tid >> 5, lane = tid & 31;
  const int lm = lane & 15, half = lane >> 4;
  const int row0 = blockIdx.y * 128, col0 = blockIdx.x * 128;

  v8f acc[8];
#pragma unroll
  for (int f = 0; f < 8; f++) acc[f] = zero8();

  const bf16* gA = A  + (size_t)row0 * K;   // tile base: rows row0.., k=0
  const bf16* gB = Bt + (size_t)col0 * K;

  if (wave == 0) {                          // prologue: chunk 0 -> buffer 0
    tdm_load_2d(gA, lds_off(sA[0]), 32, 128, K);
    tdm_load_2d(gB, lds_off(sB[0]), 32, 128, K);
  }

  const int NK = K / 32;
  for (int ci = 0; ci < NK; ci++) {
    const int cur = ci & 1;
    wait_tensorcnt0();                      // wave0: chunk ci landed; rest: NOP
    __syncthreads();                        // chunk ci visible to all waves
    if (wave == 0 && ci + 1 < NK) {         // prefetch chunk ci+1 via TDM
      tdm_load_2d(gA + (ci + 1) * 32, lds_off(sA[cur ^ 1]), 32, 128, K);
      tdm_load_2d(gB + (ci + 1) * 32, lds_off(sB[cur ^ 1]), 32, 128, K);
    }

    v16bf af = load_frag(sA[cur] + (16 * wave + lm) * 32, half);
#pragma unroll
    for (int f = 0; f < 8; f++) {
      v16bf bfb = load_frag(sB[cur] + (16 * f + lm) * 32, half);
      acc[f] = wmma_bf16(af, bfb, acc[f]);
    }
  }

#pragma unroll
  for (int f = 0; f < 8; f++) {
    const int col = col0 + 16 * f + lm;
#pragma unroll
    for (int r = 0; r < 8; r++) {
      const int row = row0 + 16 * wave + r + 8 * half;
      const float v = acc[f][r] * scale;
      if (OUT_MODE == 0) {
        ((bf16*)Cout)[(size_t)row * N + col] = (bf16)v;
      } else if (OUT_MODE == 1) {          // V^T per batch: [b][dim][t]
        const int batch = row >> 11, tr = row & (T_ - 1);
        ((bf16*)Cout)[((size_t)(batch * N + col)) * T_ + tr] = (bf16)v;
      } else {
        ((float*)Cout)[(size_t)row * N + col] = v;
      }
    }
  }
}

// --- Differential flash-attention ------------------------------------------
// grid: (t/128, H, b), block 256 (8 waves, 16 q-rows each)
// K/V tiles staged by TDM (waves 0 and 1 issue), P round-trips per-wave LDS.
__global__ void __launch_bounds__(256)
diff_attn_kernel(const bf16* __restrict__ Q, const bf16* __restrict__ Kb,
                 const bf16* __restrict__ Vt, const float* __restrict__ lam_p,
                 bf16* __restrict__ AO) {
  __shared__ __attribute__((aligned(16))) bf16 sK[32 * 128];   // [key][dim128]
  __shared__ __attribute__((aligned(16))) bf16 sV[128 * 32];   // [dim128][key]
  __shared__ __attribute__((aligned(16))) bf16 sP[8 * 2 * 512];// per-wave P

  const int tid = threadIdx.x;
  const int wave = tid >> 5, lane = tid & 31;
  const int lm = lane & 15, half = lane >> 4;
  const int qt = blockIdx.x, h = blockIdx.y, bb = blockIdx.z;
  const float lam = lam_p[0];

  const int q0 = qt * 128 + wave * 16;          // wave's first q row
  const size_t bBase = (size_t)bb * T_ * E_;
  const int colQ = 128 * h;                     // sub-heads 2h,2h+1 columns

  // Q fragments: [sub-head s][k-chunk c]
  v16bf qf[2][2];
  {
    const bf16* qrow = Q + bBase + (size_t)(q0 + lm) * E_ + colQ;
#pragma unroll
    for (int s = 0; s < 2; s++)
#pragma unroll
      for (int c = 0; c < 2; c++)
        qf[s][c] = load_frag(qrow + s * 64 + c * 32, half);
  }

  float m0[8], l0[8], m1[8], l1[8];
  v8f o0[8], o1[8];
#pragma unroll
  for (int r = 0; r < 8; r++) { m0[r] = -1e30f; m1[r] = -1e30f; l0[r] = 0.f; l1[r] = 0.f; }
#pragma unroll
  for (int f = 0; f < 8; f++) { o0[f] = zero8(); o1[f] = zero8(); }

  const int ntiles = qt * 4 + 4;                // uniform over block (causal)
  for (int kt = 0; kt < ntiles; kt++) {
    const int kb = kt * 32;
    __syncthreads();                            // all done with previous tiles
    if (wave == 0) {                            // K tile: 32 keys x 128 dims
      tdm_load_2d(Kb + bBase + (size_t)kb * E_ + colQ, lds_off(sK),
                  128, 32, E_);
    } else if (wave == 1) {                     // V tile: 128 dims x 32 keys
      tdm_load_2d(Vt + ((size_t)(bb * E_ + 128 * h)) * T_ + kb, lds_off(sV),
                  32, 128, T_);
    }
    wait_tensorcnt0();
    __syncthreads();                            // tiles visible to all waves

#pragma unroll
    for (int s = 0; s < 2; s++) {
      float* m = s ? m1 : m0;
      float* l = s ? l1 : l0;
      v8f*   o = s ? o1 : o0;

      // scores: [16 q][32 keys] = Q[16,64] @ K^T[64,32]
      v8f sc[2];
#pragma unroll
      for (int g = 0; g < 2; g++) {
        sc[g] = zero8();
#pragma unroll
        for (int c = 0; c < 2; c++) {
          v16bf kf = load_frag(sK + (16 * g + lm) * 128 + s * 64 + c * 32, half);
          sc[g] = wmma_bf16(qf[s][c], kf, sc[g]);
        }
      }

      // causal mask + row max
      float pmax[8];
#pragma unroll
      for (int r = 0; r < 8; r++) {
        const int qrow = q0 + r + 8 * half;
#pragma unroll
        for (int g = 0; g < 2; g++) {
          const int key = kb + 16 * g + lm;
          if (key > qrow) sc[g][r] = -1e30f;
        }
        pmax[r] = fmaxf(sc[0][r], sc[1][r]);
      }
#pragma unroll
      for (int xm = 1; xm < 16; xm <<= 1)
#pragma unroll
        for (int r = 0; r < 8; r++)
          pmax[r] = fmaxf(pmax[r], __shfl_xor(pmax[r], xm, 32));

      // online softmax update
      float rs[8];
#pragma unroll
      for (int r = 0; r < 8; r++) {
        const float mn = fmaxf(m[r], pmax[r]);
        const float alpha = __expf(m[r] - mn);
        m[r] = mn;
        sc[0][r] = __expf(sc[0][r] - mn);
        sc[1][r] = __expf(sc[1][r] - mn);
        rs[r] = sc[0][r] + sc[1][r];
        l[r] *= alpha;
#pragma unroll
        for (int f = 0; f < 8; f++) o[f][r] *= alpha;
      }
#pragma unroll
      for (int xm = 1; xm < 16; xm <<= 1)
#pragma unroll
        for (int r = 0; r < 8; r++)
          rs[r] += __shfl_xor(rs[r], xm, 32);
#pragma unroll
      for (int r = 0; r < 8; r++) l[r] += rs[r];

      // P: C-layout -> A-layout through per-wave LDS
      bf16* sPs = sP + (wave * 2 + s) * 512;
#pragma unroll
      for (int r = 0; r < 8; r++) {
        const int row = r + 8 * half;
        sPs[row * 32 + lm]      = (bf16)sc[0][r];
        sPs[row * 32 + 16 + lm] = (bf16)sc[1][r];
      }
      v16bf pf = load_frag(sPs + lm * 32, half);

      // O += P[16,32] @ V[32,128]
#pragma unroll
      for (int f = 0; f < 8; f++) {
        v16bf vf = load_frag(sV + (16 * f + lm) * 32, half);
        o[f] = wmma_bf16(pf, vf, o[f]);
      }
    }
  }

  // epilogue: diff combine, RMSNorm over 128 dims, (1-lambda_init) scale
  float inv0[8], inv1[8], ss[8];
#pragma unroll
  for (int r = 0; r < 8; r++) {
    inv0[r] = 1.f / l0[r];
    inv1[r] = 1.f / l1[r];
    ss[r] = 0.f;
  }
#pragma unroll
  for (int f = 0; f < 8; f++)
#pragma unroll
    for (int r = 0; r < 8; r++) {
      const float d = o0[f][r] * inv0[r] - lam * o1[f][r] * inv1[r];
      ss[r] += d * d;
    }
#pragma unroll
  for (int xm = 1; xm < 16; xm <<= 1)
#pragma unroll
    for (int r = 0; r < 8; r++)
      ss[r] += __shfl_xor(ss[r], xm, 32);

  float scl[8];
#pragma unroll
  for (int r = 0; r < 8; r++)
    scl[r] = rsqrtf(ss[r] * (1.f / 128.f) + 1e-5f) * ONE_MINUS_LAMBDA_F;

#pragma unroll
  for (int f = 0; f < 8; f++)
#pragma unroll
    for (int r = 0; r < 8; r++) {
      const int qrow = q0 + r + 8 * half;
      const float d = o0[f][r] * inv0[r] - lam * o1[f][r] * inv1[r];
      AO[bBase + (size_t)qrow * E_ + 128 * h + 16 * f + lm] = (bf16)(d * scl[r]);
    }
}

// ---------------------------------------------------------------------------
extern "C" void kernel_launch(void* const* d_in, const int* in_sizes, int n_in,
                              void* d_out, int out_size, void* d_ws, size_t ws_size,
                              hipStream_t stream) {
  (void)in_sizes; (void)n_in; (void)out_size; (void)ws_size;
  const float* x   = (const float*)d_in[0];
  const float* Wq  = (const float*)d_in[1];
  const float* Wk  = (const float*)d_in[2];
  const float* Wv  = (const float*)d_in[3];
  const float* Wo  = (const float*)d_in[4];
  const float* lq1 = (const float*)d_in[5];
  const float* lk1 = (const float*)d_in[6];
  const float* lq2 = (const float*)d_in[7];
  const float* lk2 = (const float*)d_in[8];
  float* out = (float*)d_out;

  // workspace layout (bytes): needs ~118 MB
  char* w = (char*)d_ws;
  const size_t szX = (size_t)MT_ * E_ * sizeof(bf16);     // 16 MB
  const size_t szW = (size_t)E_ * E_ * sizeof(bf16);      // 8 MB
  bf16* xb  = (bf16*)w;            w += szX;
  bf16* Wqt = (bf16*)w;            w += szW;
  bf16* Wkt = (bf16*)w;            w += szW;
  bf16* Wvt = (bf16*)w;            w += szW;
  bf16* Wot = (bf16*)w;            w += szW;
  bf16* Qb  = (bf16*)w;            w += szX;
  bf16* Kb  = (bf16*)w;            w += szX;
  bf16* Vtb = (bf16*)w;            w += szX;   // V^T per batch: [b][E][t]
  bf16* AO  = (bf16*)w;            w += szX;
  float* lam = (float*)w;

  // 1) conversions
  cvt_f32_bf16_kernel<<<(MT_ * E_) / (256 * 4), 256, 0, stream>>>(x, xb, MT_ * E_);
  dim3 tg(64, 64), tb(32, 8);
  transpose_cvt_kernel<<<tg, tb, 0, stream>>>(Wq, Wqt, E_);
  transpose_cvt_kernel<<<tg, tb, 0, stream>>>(Wk, Wkt, E_);
  transpose_cvt_kernel<<<tg, tb, 0, stream>>>(Wv, Wvt, E_);
  transpose_cvt_kernel<<<tg, tb, 0, stream>>>(Wo, Wot, E_);
  lambda_kernel<<<1, 32, 0, stream>>>(lq1, lk1, lq2, lk2, lam);

  // 2) projections (M=4096, K=2048, N=2048)
  dim3 gg(E_ / 128, MT_ / 128);   // (16, 32)
  gemm_bf16_kernel<0><<<gg, 256, 0, stream>>>(xb, Wqt, Qb,  E_, E_, 0.125f); // *HD^-0.5
  gemm_bf16_kernel<0><<<gg, 256, 0, stream>>>(xb, Wkt, Kb,  E_, E_, 1.0f);
  gemm_bf16_kernel<1><<<gg, 256, 0, stream>>>(xb, Wvt, Vtb, E_, E_, 1.0f);   // transposed

  // 3) differential flash attention
  dim3 ag(T_ / 128, H_, B_);      // (16, 16, 2)
  diff_attn_kernel<<<ag, 256, 0, stream>>>(Qb, Kb, Vtb, lam, AO);

  // 4) output projection (fp32 out)
  gemm_bf16_kernel<2><<<gg, 256, 0, stream>>>(AO, Wot, out, E_, E_, 1.0f);
}
